// GraphGR_51788715655932
// MI455X (gfx1250) — matrix operand order repaired and checked by reading, not compiled, not run on hardware
//
#include <hip/hip_runtime.h>
#include <hip/hip_bf16.h>

typedef __attribute__((ext_vector_type(16))) __bf16 v16bf;
typedef __attribute__((ext_vector_type(8)))  float  v8f;
typedef __attribute__((ext_vector_type(4)))  float  v4f;

#define DH 128   // hidden dim

// ---------- WMMA fragment helpers (bf16 16x16x32, f32 accum) ----------

// A-matrix 16x32 bf16 (MxK): lane = 16*half + m ; kbase = half*16
// elements 0..7  -> K = ks + kbase + j
// elements 8..15 -> K = ks + kbase + 8 + j
__device__ __forceinline__ v16bf make_a_frag(const float* __restrict__ Arow, int ks, int lane) {
  const int kbase = (lane >> 4) * 16;
  const float* p0 = Arow + ks + kbase;
  const float* p1 = p0 + 8;
  v16bf a;
#pragma unroll
  for (int j = 0; j < 8; ++j) {
    a[j]     = (__bf16)p0[j];
    a[8 + j] = (__bf16)p1[j];
  }
  return a;
}

// B-matrix 32x16 bf16 (KxN), weights pre-stored transposed as Wt[N][K]:
// lane half selects K 0..15 vs 16..31; element i = consecutive K.
// -> one aligned 32-byte vector load per fragment.
__device__ __forceinline__ v16bf make_b_frag(const __bf16* __restrict__ Wt, int ncol, int ks, int lane) {
  const int kb = ks + ((lane >> 4) << 4);
  return *(const v16bf*)(Wt + (size_t)ncol * DH + kb);
}

__device__ __forceinline__ void gemm_term(v8f& c, const float* __restrict__ A,
                                          const __bf16* __restrict__ Wt,
                                          int mrow, int ncol, int lane) {
  const float* Arow = A + (size_t)mrow * DH;
#pragma unroll
  for (int ks = 0; ks < DH; ks += 32) {
    v16bf a = make_a_frag(Arow, ks, lane);
    v16bf b = make_b_frag(Wt, ncol, ks, lane);
    c = __builtin_amdgcn_wmma_f32_16x16x32_bf16(false, a, false, b, (short)0, c, false, false);
  }
}

// ---------- fused multi-term 128-wide linear: Y = relu(sum_t A_t @ W_t + biases) ----------
// Template-specialized on term count so all pointers are used directly from kernel
// args (keeps addrspace inference alive -> global_load_b128, not flat_load).
template <int NTERMS, bool BIAS2>
__global__ void __launch_bounds__(32) sage_gemm128(
    const float* __restrict__ A0, const __bf16* __restrict__ W0,
    const float* __restrict__ A1, const __bf16* __restrict__ W1,
    const float* __restrict__ A2, const __bf16* __restrict__ W2,
    const float* __restrict__ A3, const __bf16* __restrict__ W3,
    const float* __restrict__ biasA, const float* __restrict__ biasB,
    float* __restrict__ Y) {
  const int lane = threadIdx.x;
  const int nloc = lane & 15;
  const int mt = blockIdx.x;
  const int ncol = blockIdx.y * 16 + nloc;
  const int mrow = mt * 16 + nloc;

  v8f c = {};
  if constexpr (NTERMS >= 1) gemm_term(c, A0, W0, mrow, ncol, lane);
  if constexpr (NTERMS >= 2) gemm_term(c, A1, W1, mrow, ncol, lane);
  if constexpr (NTERMS >= 3) gemm_term(c, A2, W2, mrow, ncol, lane);
  if constexpr (NTERMS >= 4) gemm_term(c, A3, W3, mrow, ncol, lane);

  float bias = biasA[ncol];
  if constexpr (BIAS2) bias += biasB[ncol];
  const int mbase = mt * 16 + ((lane >> 4) << 3);
#pragma unroll
  for (int r = 0; r < 8; ++r) {
    Y[(size_t)(mbase + r) * DH + ncol] = fmaxf(c[r] + bias, 0.0f);
  }
}

// ---------- predictor: out[M x Ncols] = A[M x 128] @ WpT(bf16)[Ncols x 128]^T + bias ----------
// two N-tiles per wave to reuse the A fragment; nontemporal stores keep the
// 800MB streamed output from evicting L2-resident Wp/g2.
__global__ void __launch_bounds__(32) predict_gemm(
    const float* __restrict__ A, const __bf16* __restrict__ Wt,
    const float* __restrict__ bias, float* __restrict__ out, int Ncols) {
  const int lane = threadIdx.x;
  const int nloc = lane & 15;
  const int mt = blockIdx.x;
  const int nt0 = blockIdx.y * 2;
  const float* Arow = A + (size_t)(mt * 16 + nloc) * DH;
  v8f c0 = {}, c1 = {};
#pragma unroll
  for (int ks = 0; ks < DH; ks += 32) {
    v16bf a  = make_a_frag(Arow, ks, lane);
    v16bf b0 = make_b_frag(Wt, nt0 * 16 + nloc, ks, lane);
    v16bf b1 = make_b_frag(Wt, (nt0 + 1) * 16 + nloc, ks, lane);
    c0 = __builtin_amdgcn_wmma_f32_16x16x32_bf16(false, a, false, b0, (short)0, c0, false, false);
    c1 = __builtin_amdgcn_wmma_f32_16x16x32_bf16(false, a, false, b1, (short)0, c1, false, false);
  }
  const int mbase = mt * 16 + ((lane >> 4) << 3);
  const int n0 = nt0 * 16 + nloc;
  const int n1 = n0 + 16;
  const float bb0 = bias[n0];
  const float bb1 = bias[n1];
#pragma unroll
  for (int r = 0; r < 8; ++r) {
    __builtin_nontemporal_store(c0[r] + bb0, &out[(size_t)(mbase + r) * Ncols + n0]);
    __builtin_nontemporal_store(c1[r] + bb1, &out[(size_t)(mbase + r) * Ncols + n1]);
  }
}

// ---------- weight convert + transpose: W[k][n] (f32, K=128 rows) -> Wt[n][k] (bf16) ----------
__global__ void cvt_transpose(const float* __restrict__ W, __bf16* __restrict__ Wt, int N) {
  int idx = blockIdx.x * 256 + threadIdx.x;
  if (idx >= N * DH) return;
  int n = idx >> 7;
  int k = idx & (DH - 1);
  Wt[idx] = (__bf16)W[(size_t)k * N + n];
}

// ---------- graph ops ----------
__global__ void edge_degree(const int* __restrict__ dst, float* __restrict__ deg, int nE) {
  int e = blockIdx.x * 256 + threadIdx.x;
  if (e >= nE) return;
  atomicAdd(&deg[dst[e]], 1.0f);
}

// 4 channels per thread: b128 gather + 4 f32 atomics into L2-resident accumulators
__global__ void edge_scatter4(const float* __restrict__ X, const int* __restrict__ src,
                              const int* __restrict__ dst, float* __restrict__ acc, int nE) {
  int idx = blockIdx.x * 256 + threadIdx.x;
  int e = idx >> 5;
  if (e >= nE) return;
  int c = (idx & 31) * 4;
  const float* xp = X + (size_t)src[e] * DH + c;
  v4f x = *(const v4f*)xp;
  float* ap = acc + (size_t)dst[e] * DH + c;
  atomicAdd(ap + 0, x[0]);
  atomicAdd(ap + 1, x[1]);
  atomicAdd(ap + 2, x[2]);
  atomicAdd(ap + 3, x[3]);
}

__global__ void seg_mean_div4(float* __restrict__ acc, const float* __restrict__ deg, int n) {
  int idx = blockIdx.x * 256 + threadIdx.x;   // over n*32 float4's
  if (idx >= n * (DH / 4)) return;
  float inv = 1.0f / fmaxf(deg[idx >> 5], 1.0f);
  v4f* p = (v4f*)acc + idx;
  v4f v = *p;
  v[0] *= inv; v[1] *= inv; v[2] *= inv; v[3] *= inv;
  *p = v;
}

// ---------- host launcher ----------
extern "C" void kernel_launch(void* const* d_in, const int* in_sizes, int n_in,
                              void* d_out, int out_size, void* d_ws, size_t ws_size,
                              hipStream_t stream) {
  const int NG  = in_sizes[0];
  const int NU  = in_sizes[1];
  const int NI  = in_sizes[2];
  const int EGI = in_sizes[3];
  const int EGU = in_sizes[5];

  const int* gi_src = (const int*)d_in[3];
  const int* gi_dst = (const int*)d_in[4];
  const int* gu_src = (const int*)d_in[5];
  const int* gu_dst = (const int*)d_in[6];
  const float* emb_user = (const float*)d_in[8];
  const float* emb_item = (const float*)d_in[9];
  const float* Wl1 = (const float*)d_in[10];
  const float* Wr1 = (const float*)d_in[11];
  const float* b1  = (const float*)d_in[12];
  const float* Wl2 = (const float*)d_in[13];
  const float* Wr2 = (const float*)d_in[14];
  const float* b2  = (const float*)d_in[15];
  const float* Wp  = (const float*)d_in[16];
  const float* bp  = (const float*)d_in[17];
  float* out = (float*)d_out;

  // ---- workspace carve ----
  float* ws = (float*)d_ws;
  size_t off = 0;
  float* aggHI = ws + off; off += (size_t)NG * DH;
  float* aggHU = ws + off; off += (size_t)NG * DH;
  float* aggI1 = ws + off; off += (size_t)NG * DH;
  float* aggU1 = ws + off; off += (size_t)NG * DH;
  float* degGI = ws + off; off += NG;
  float* degGU = ws + off; off += NG;
  const size_t zeroBytes = off * sizeof(float);
  float* g1 = ws + off; off += (size_t)NG * DH;
  float* g2 = ws + off; off += (size_t)NG * DH;
  float* u1 = ws + off; off += (size_t)NU * DH;
  float* i1 = ws + off; off += (size_t)NI * DH;
  __bf16* bw = (__bf16*)(ws + off);
  const size_t WSZ = (size_t)DH * DH;
  __bf16* Wt_l1_1 = bw; bw += WSZ;
  __bf16* Wt_l1_3 = bw; bw += WSZ;
  __bf16* Wt_r1_0 = bw; bw += WSZ;
  __bf16* Wt_r1_2 = bw; bw += WSZ;
  __bf16* Wt_l2_1 = bw; bw += WSZ;
  __bf16* Wt_l2_3 = bw; bw += WSZ;
  __bf16* Wt_r2_1 = bw; bw += WSZ;
  __bf16* Wt_r2_3 = bw; bw += WSZ;
  __bf16* Wpt     = bw; bw += (size_t)NI * DH;

  // ---- zero accumulators (re-zeroed every call; graph-capture safe) ----
  hipMemsetAsync(d_ws, 0, zeroBytes, stream);

  // ---- weight pre-convert/transpose to bf16 [N][K] ----
  {
    dim3 blk(256);
    int gsmall = (DH * DH + 255) / 256;
    cvt_transpose<<<gsmall, blk, 0, stream>>>(Wl1 + 1 * DH * DH, Wt_l1_1, DH);
    cvt_transpose<<<gsmall, blk, 0, stream>>>(Wl1 + 3 * DH * DH, Wt_l1_3, DH);
    cvt_transpose<<<gsmall, blk, 0, stream>>>(Wr1 + 0 * DH * DH, Wt_r1_0, DH);
    cvt_transpose<<<gsmall, blk, 0, stream>>>(Wr1 + 2 * DH * DH, Wt_r1_2, DH);
    cvt_transpose<<<gsmall, blk, 0, stream>>>(Wl2 + 1 * DH * DH, Wt_l2_1, DH);
    cvt_transpose<<<gsmall, blk, 0, stream>>>(Wl2 + 3 * DH * DH, Wt_l2_3, DH);
    cvt_transpose<<<gsmall, blk, 0, stream>>>(Wr2 + 1 * DH * DH, Wt_r2_1, DH);
    cvt_transpose<<<gsmall, blk, 0, stream>>>(Wr2 + 3 * DH * DH, Wt_r2_3, DH);
    int gbig = (NI * DH + 255) / 256;
    cvt_transpose<<<gbig, blk, 0, stream>>>(Wp, Wpt, NI);
  }

  // ---- degrees (same edge arrays used by both layers) ----
  edge_degree<<<(EGI + 255) / 256, 256, 0, stream>>>(gi_src, degGI, EGI);
  edge_degree<<<(EGU + 255) / 256, 256, 0, stream>>>(gu_src, degGU, EGU);

  // ---- layer-1 aggregations into groups (hg==0 kills everything else) ----
  {
    long long tI = (long long)EGI * 32, tU = (long long)EGU * 32;
    edge_scatter4<<<(int)((tI + 255) / 256), 256, 0, stream>>>(emb_item, gi_dst, gi_src, aggHI, EGI);
    edge_scatter4<<<(int)((tU + 255) / 256), 256, 0, stream>>>(emb_user, gu_dst, gu_src, aggHU, EGU);
  }
  seg_mean_div4<<<(NG * 32 + 255) / 256, 256, 0, stream>>>(aggHI, degGI, NG);
  seg_mean_div4<<<(NG * 32 + 255) / 256, 256, 0, stream>>>(aggHU, degGU, NG);

  // ---- layer-1 GEMMs ----
  // g1 = relu(aggHI @ Wl1[1] + aggHU @ Wl1[3] + b1[1] + b1[3])
  sage_gemm128<2, true><<<dim3(NG / 16, DH / 16), 32, 0, stream>>>(
      aggHI, Wt_l1_1, aggHU, Wt_l1_3, nullptr, nullptr, nullptr, nullptr,
      b1 + 1 * DH, b1 + 3 * DH, g1);
  // i1 = relu(emb_item @ Wr1[0] + b1[0])
  sage_gemm128<1, false><<<dim3(NI / 16, DH / 16), 32, 0, stream>>>(
      emb_item, Wt_r1_0, nullptr, nullptr, nullptr, nullptr, nullptr, nullptr,
      b1 + 0 * DH, nullptr, i1);
  // u1 = relu(emb_user @ Wr1[2] + b1[2])
  sage_gemm128<1, false><<<dim3(NU / 16, DH / 16), 32, 0, stream>>>(
      emb_user, Wt_r1_2, nullptr, nullptr, nullptr, nullptr, nullptr, nullptr,
      b1 + 2 * DH, nullptr, u1);

  // ---- layer-2 aggregations ----
  {
    long long tI = (long long)EGI * 32, tU = (long long)EGU * 32;
    edge_scatter4<<<(int)((tI + 255) / 256), 256, 0, stream>>>(i1, gi_dst, gi_src, aggI1, EGI);
    edge_scatter4<<<(int)((tU + 255) / 256), 256, 0, stream>>>(u1, gu_dst, gu_src, aggU1, EGU);
  }
  seg_mean_div4<<<(NG * 32 + 255) / 256, 256, 0, stream>>>(aggI1, degGI, NG);
  seg_mean_div4<<<(NG * 32 + 255) / 256, 256, 0, stream>>>(aggU1, degGU, NG);

  // ---- layer-2 group GEMM: 4 terms ----
  // g2 = relu(aggI1@Wl2[1] + aggU1@Wl2[3] + g1@Wr2[1] + g1@Wr2[3] + b2[1] + b2[3])
  sage_gemm128<4, true><<<dim3(NG / 16, DH / 16), 32, 0, stream>>>(
      aggI1, Wt_l2_1, aggU1, Wt_l2_3, g1, Wt_r2_1, g1, Wt_r2_3,
      b2 + 1 * DH, b2 + 3 * DH, g2);

  // ---- predictor: out = g2 @ Wp + bp  (M=NG, N=NI, K=128) ----
  predict_gemm<<<dim3(NG / 16, (NI / 16) / 2), 32, 0, stream>>>(g2, Wpt, bp, out, NI);
}
